// LatentODE_19791209300720
// MI455X (gfx1250) — matrix operand order repaired and checked by reading, not compiled, not run on hardware
//
#include <hip/hip_runtime.h>
#include <hip/hip_bf16.h>
#include <math.h>

// Problem sizes (fixed by the reference)
#define BB   2048
#define TT   256
#define NVV  32
#define HH   64
#define LL   32
#define OHH  64
#define ODE_STEPS 64
#define ODE_TEND  48.0f

typedef __attribute__((ext_vector_type(8)))  float  v8f;
typedef __attribute__((ext_vector_type(16))) __bf16 v16bf;
typedef __attribute__((ext_vector_type(8)))  __bf16 v8bf;

static __device__ __forceinline__ v8f vzero8() {
  v8f z;
#pragma unroll
  for (int i = 0; i < 8; ++i) z[i] = 0.0f;
  return z;
}

static __device__ __forceinline__ v8f wmma_bf16(v16bf a, v16bf b, v8f c) {
  // D = A(16x32 bf16) * B(32x16 bf16) + C(16x16 f32)
  return __builtin_amdgcn_wmma_f32_16x16x32_bf16(false, a, false, b, (short)0, c,
                                                 false, false);
}



static __device__ __forceinline__ v16bf cat8(v8bf a, v8bf b) {
  v16bf r;
#pragma unroll
  for (int i = 0; i < 8; ++i) { r[i] = a[i]; r[8 + i] = b[i]; }
  return r;
}

// B fragment: W is row-major [N_total][K_stride] bf16; lane holds N = g,
// elements i -> K = koff + (i&7) + 16*(i>>3) + 8*hi : two 16B runs.
static __device__ __forceinline__ v16bf bfrag(const __bf16* W, int stride, int g,
                                              int koff, int hi) {
  v8bf a = *(const v8bf*)(W + g * stride + koff + 8 * hi);
  v8bf b = *(const v8bf*)(W + g * stride + koff + 16 + 8 * hi);
  return cat8(a, b);
}

// A fragment from row-major bf16 [16][stride]; lane row = col (lane&15),
// elements i -> K = koff + (i&7) + 16*(i>>3) + 8*hi : two 16B runs.
static __device__ __forceinline__ v16bf afrag_lds(const __bf16* S, int stride,
                                                  int row, int koff, int hi) {
  v8bf a = *(const v8bf*)(S + row * stride + koff + 8 * hi);
  v8bf b = *(const v8bf*)(S + row * stride + koff + 16 + 8 * hi);
  return cat8(a, b);
}

// A fragment (one K=32 chunk) from a global row-major f32 row pointer.
static __device__ __forceinline__ v16bf afrag_gbl_f32(const float* p, int hi) {
  const float4 a0 = *(const float4*)(p + 8 * hi);
  const float4 a1 = *(const float4*)(p + 8 * hi + 4);
  const float4 b0 = *(const float4*)(p + 16 + 8 * hi);
  const float4 b1 = *(const float4*)(p + 16 + 8 * hi + 4);
  v16bf r;
  r[0] = (__bf16)a0.x;  r[1] = (__bf16)a0.y;  r[2] = (__bf16)a0.z;  r[3] = (__bf16)a0.w;
  r[4] = (__bf16)a1.x;  r[5] = (__bf16)a1.y;  r[6] = (__bf16)a1.z;  r[7] = (__bf16)a1.w;
  r[8] = (__bf16)b0.x;  r[9] = (__bf16)b0.y;  r[10] = (__bf16)b0.z; r[11] = (__bf16)b0.w;
  r[12] = (__bf16)b1.x; r[13] = (__bf16)b1.y; r[14] = (__bf16)b1.z; r[15] = (__bf16)b1.w;
  return r;
}

// Stage a set of D-layout f32 tiles into a row-major [16][KT] bf16 LDS buffer.
// D layout: element e of lane L -> row M = e + 8*hi, col N = (L&15) + 16*q.
template <int NT, int KT>
static __device__ __forceinline__ void stageD(__bf16* S, const v8f* t, int col, int hi) {
#pragma unroll
  for (int q = 0; q < NT; ++q)
#pragma unroll
    for (int e = 0; e < 8; ++e)
      S[(e + 8 * hi) * KT + col + 16 * q] = (__bf16)t[q][e];
}

// One evaluation of the latent ODE MLP: fo = W3 * tanh(W2 * tanh(W1*z+b1)+b2)+b3
static __device__ __forceinline__ void ode_f(const v8f zin[2], v8f fo[2], __bf16* S,
                                             const __bf16* sW1, const __bf16* sW2,
                                             const __bf16* sW3, const float* sB1,
                                             const float* sB2, const float* sB3,
                                             int col, int hi) {
  __syncthreads();
  stageD<2, 32>(S, zin, col, hi);
  __syncthreads();
  v16bf zA = afrag_lds(S, 32, col, 0, hi);
  v8f h1[4];
#pragma unroll
  for (int q = 0; q < 4; ++q) {
    v8f d = wmma_bf16(zA, bfrag(sW1, 32, q * 16 + col, 0, hi), vzero8());
    float b = sB1[col + 16 * q];
#pragma unroll
    for (int e = 0; e < 8; ++e) h1[q][e] = tanhf(d[e] + b);
  }
  __syncthreads();
  stageD<4, 64>(S, h1, col, hi);
  __syncthreads();
  v16bf a0 = afrag_lds(S, 64, col, 0, hi);
  v16bf a1 = afrag_lds(S, 64, col, 32, hi);
  v8f h2[4];
#pragma unroll
  for (int q = 0; q < 4; ++q) {
    int g = q * 16 + col;
    v8f d = wmma_bf16(a0, bfrag(sW2, 64, g, 0, hi), vzero8());
    d = wmma_bf16(a1, bfrag(sW2, 64, g, 32, hi), d);
    float b = sB2[col + 16 * q];
#pragma unroll
    for (int e = 0; e < 8; ++e) h2[q][e] = tanhf(d[e] + b);
  }
  __syncthreads();
  stageD<4, 64>(S, h2, col, hi);
  __syncthreads();
  v16bf c0 = afrag_lds(S, 64, col, 0, hi);
  v16bf c1 = afrag_lds(S, 64, col, 32, hi);
#pragma unroll
  for (int q = 0; q < 2; ++q) {
    int g = q * 16 + col;
    v8f d = wmma_bf16(c0, bfrag(sW3, 64, g, 0, hi), vzero8());
    d = wmma_bf16(c1, bfrag(sW3, 64, g, 32, hi), d);
    float b = sB3[col + 16 * q];
#pragma unroll
    for (int e = 0; e < 8; ++e) fo[q][e] = d[e] + b;
  }
}

__global__ void latent_ode_init_kernel(float* out) {
  if (threadIdx.x == 0) out[BB] = 0.0f;  // KL accumulator slot
}

__global__ __launch_bounds__(32) void latent_ode_kernel(
    const float* __restrict__ values, const float* __restrict__ maskp,
    const int* __restrict__ seqlen, const float* __restrict__ eps,
    const float* __restrict__ W_ih, const float* __restrict__ W_hh,
    const float* __restrict__ b_ih, const float* __restrict__ b_hh,
    const float* __restrict__ W_z0, const float* __restrict__ b_z0,
    const float* __restrict__ oW1, const float* __restrict__ ob1,
    const float* __restrict__ oW2, const float* __restrict__ ob2,
    const float* __restrict__ oW3, const float* __restrict__ ob3,
    const float* __restrict__ dW1, const float* __restrict__ db1,
    const float* __restrict__ dW2, const float* __restrict__ db2,
    float* __restrict__ out) {
  __shared__ __bf16 sWih[192 * 64];
  __shared__ __bf16 sWhh[192 * 64];
  __shared__ __bf16 sWz0[64 * 64];
  __shared__ __bf16 sW1[OHH * LL];
  __shared__ __bf16 sW2[OHH * OHH];
  __shared__ __bf16 sW3[LL * OHH];
  __shared__ __bf16 sDW1[HH * LL];
  __shared__ float sBih[192], sBhh[192], sBz0[64], sB1[OHH], sB2[OHH], sB3[LL];
  __shared__ float sDb1[HH], sDw2[HH];
  __shared__ __bf16 sStage[16 * 64];
  __shared__ float sStageF[16 * 64];

  const int lane = threadIdx.x;
  const int col = lane & 15;
  const int hi = lane >> 4;
  const int b0 = blockIdx.x * 16;

  // ---- prologue: convert weights to bf16 in LDS ----
  for (int i = lane; i < 192 * 64; i += 32) {
    sWih[i] = (__bf16)W_ih[i];
    sWhh[i] = (__bf16)W_hh[i];
  }
  for (int i = lane; i < 64 * 64; i += 32) {
    sWz0[i] = (__bf16)W_z0[i];
    sW2[i] = (__bf16)oW2[i];
  }
  for (int i = lane; i < 64 * 32; i += 32) {
    sW1[i] = (__bf16)oW1[i];
    sDW1[i] = (__bf16)dW1[i];
    sW3[i] = (__bf16)oW3[i];
  }
  for (int i = lane; i < 192; i += 32) {
    sBih[i] = b_ih[i];
    sBhh[i] = b_hh[i];
  }
  for (int i = lane; i < 64; i += 32) {
    sBz0[i] = b_z0[i];
    sB1[i] = ob1[i];
    sB2[i] = ob2[i];
    sDb1[i] = db1[i];
    sDw2[i] = dW2[i];
  }
  for (int i = lane; i < 32; i += 32) sB3[i] = ob3[i];
  const float db2v = db2[0];
  __syncthreads();

  // per-lane gate biases
  float bR[4], bZ[4], bXn[4], bHn[4];
#pragma unroll
  for (int q = 0; q < 4; ++q) {
    int g = q * 16 + col;
    bR[q] = sBih[g] + sBhh[g];
    bZ[q] = sBih[64 + g] + sBhh[64 + g];
    bXn[q] = sBih[128 + g];
    bHn[q] = sBhh[128 + g];
  }
  int sl[8];
#pragma unroll
  for (int e = 0; e < 8; ++e) sl[e] = seqlen[b0 + e + 8 * hi];

  // ---- GRU scan over reversed time ----
  v8f hD[4];
#pragma unroll
  for (int q = 0; q < 4; ++q) hD[q] = vzero8();

#pragma unroll 1
  for (int t = 0; t < TT; ++t) {
    stageD<4, 64>(sStage, hD, col, hi);
    __syncthreads();
    v16bf hA0 = afrag_lds(sStage, 64, col, 0, hi);
    v16bf hA1 = afrag_lds(sStage, 64, col, 32, hi);

    const int tm = TT - 1 - t;
    const float* vrow = values + ((size_t)(b0 + col) * TT + tm) * NVV;
    const float* mrow = maskp + ((size_t)(b0 + col) * TT + tm) * NVV;
    v16bf xA0 = afrag_gbl_f32(vrow, hi);
    v16bf xA1 = afrag_gbl_f32(mrow, hi);

    v8f accRZ[8], accXn[4], accHn[4];
#pragma unroll
    for (int n = 0; n < 8; ++n) {
      int g = n * 16 + col;
      v8f a = wmma_bf16(xA0, bfrag(sWih, 64, g, 0, hi), vzero8());
      a = wmma_bf16(xA1, bfrag(sWih, 64, g, 32, hi), a);
      a = wmma_bf16(hA0, bfrag(sWhh, 64, g, 0, hi), a);
      a = wmma_bf16(hA1, bfrag(sWhh, 64, g, 32, hi), a);
      accRZ[n] = a;
    }
#pragma unroll
    for (int n = 0; n < 4; ++n) {
      int g = (8 + n) * 16 + col;
      v8f ax = wmma_bf16(xA0, bfrag(sWih, 64, g, 0, hi), vzero8());
      ax = wmma_bf16(xA1, bfrag(sWih, 64, g, 32, hi), ax);
      accXn[n] = ax;
      v8f ah = wmma_bf16(hA0, bfrag(sWhh, 64, g, 0, hi), vzero8());
      ah = wmma_bf16(hA1, bfrag(sWhh, 64, g, 32, hi), ah);
      accHn[n] = ah;
    }
#pragma unroll
    for (int q = 0; q < 4; ++q) {
#pragma unroll
      for (int e = 0; e < 8; ++e) {
        float rg = 1.0f / (1.0f + expf(-(accRZ[q][e] + bR[q])));
        float zg = 1.0f / (1.0f + expf(-(accRZ[4 + q][e] + bZ[q])));
        float ng = tanhf(accXn[q][e] + bXn[q] + rg * (accHn[q][e] + bHn[q]));
        float hn = (1.0f - zg) * ng + zg * hD[q][e];
        hD[q][e] = (t < sl[e]) ? hn : hD[q][e];
      }
    }
    __syncthreads();
  }

  // ---- z0 projection + reparameterization + KL ----
  stageD<4, 64>(sStage, hD, col, hi);
  __syncthreads();
  {
    v16bf hA0 = afrag_lds(sStage, 64, col, 0, hi);
    v16bf hA1 = afrag_lds(sStage, 64, col, 32, hi);
    v8f zp[4];
#pragma unroll
    for (int q = 0; q < 4; ++q) {
      int g = q * 16 + col;
      v8f d = wmma_bf16(hA0, bfrag(sWz0, 64, g, 0, hi), vzero8());
      d = wmma_bf16(hA1, bfrag(sWz0, 64, g, 32, hi), d);
      float b = sBz0[col + 16 * q];
#pragma unroll
      for (int e = 0; e < 8; ++e) zp[q][e] = d[e] + b;
    }
    v8f zD[2];
    float kls = 0.0f;
#pragma unroll
    for (int q = 0; q < 2; ++q) {
#pragma unroll
      for (int e = 0; e < 8; ++e) {
        float m = zp[q][e];
        float lv = zp[q + 2][e];
        float ev = eps[(size_t)(b0 + e + 8 * hi) * LL + col + 16 * q];
        zD[q][e] = m + ev * expf(0.5f * lv);
        kls += 1.0f + lv - m * m - expf(lv);
      }
    }
    // wave32 reduction, then one atomic per block
    for (int m = 16; m >= 1; m >>= 1) kls += __shfl_xor(kls, m, 32);
    if (lane == 0)
      atomicAdd(out + BB, kls * (-0.5f / (float)(BB * LL)));

    // ---- fixed-step RK4 latent ODE integration to t = 48 ----
    const float dt = ODE_TEND / (float)ODE_STEPS;
    v8f k1[2], k2[2], k3[2], k4[2], zt[2];
#pragma unroll 1
    for (int s = 0; s < ODE_STEPS; ++s) {
      ode_f(zD, k1, sStage, sW1, sW2, sW3, sB1, sB2, sB3, col, hi);
#pragma unroll
      for (int q = 0; q < 2; ++q)
#pragma unroll
        for (int e = 0; e < 8; ++e) zt[q][e] = zD[q][e] + 0.5f * dt * k1[q][e];
      ode_f(zt, k2, sStage, sW1, sW2, sW3, sB1, sB2, sB3, col, hi);
#pragma unroll
      for (int q = 0; q < 2; ++q)
#pragma unroll
        for (int e = 0; e < 8; ++e) zt[q][e] = zD[q][e] + 0.5f * dt * k2[q][e];
      ode_f(zt, k3, sStage, sW1, sW2, sW3, sB1, sB2, sB3, col, hi);
#pragma unroll
      for (int q = 0; q < 2; ++q)
#pragma unroll
        for (int e = 0; e < 8; ++e) zt[q][e] = zD[q][e] + dt * k3[q][e];
      ode_f(zt, k4, sStage, sW1, sW2, sW3, sB1, sB2, sB3, col, hi);
#pragma unroll
      for (int q = 0; q < 2; ++q)
#pragma unroll
        for (int e = 0; e < 8; ++e)
          zD[q][e] += (dt / 6.0f) *
                      (k1[q][e] + 2.0f * k2[q][e] + 2.0f * k3[q][e] + k4[q][e]);
    }

    // ---- decoder: logits = relu(z @ dW1^T + db1) @ dW2^T + db2 ----
    __syncthreads();
    stageD<2, 32>(sStage, zD, col, hi);
    __syncthreads();
    v16bf zA = afrag_lds(sStage, 32, col, 0, hi);
#pragma unroll
    for (int q = 0; q < 4; ++q) {
      v8f d = wmma_bf16(zA, bfrag(sDW1, 32, q * 16 + col, 0, hi), vzero8());
      float b = sDb1[col + 16 * q];
#pragma unroll
      for (int e = 0; e < 8; ++e)
        sStageF[(e + 8 * hi) * 64 + col + 16 * q] = fmaxf(d[e] + b, 0.0f);
    }
    __syncthreads();
    if (lane < 16) {
      float s = 0.0f;
#pragma unroll 1
      for (int j = 0; j < 64; ++j) s += sStageF[lane * 64 + j] * sDw2[j];
      out[b0 + lane] = s + db2v;
    }
  }
}

extern "C" void kernel_launch(void* const* d_in, const int* in_sizes, int n_in,
                              void* d_out, int out_size, void* d_ws, size_t ws_size,
                              hipStream_t stream) {
  (void)in_sizes; (void)n_in; (void)d_ws; (void)ws_size; (void)out_size;
  // setup_inputs order:
  // 0 times, 1 values, 2 mask, 3 seq_lengths, 4 eps, 5 W_ih, 6 W_hh, 7 b_ih,
  // 8 b_hh, 9 W_z0, 10 b_z0, 11 ode_W1, 12 ode_b1, 13 ode_W2, 14 ode_b2,
  // 15 ode_W3, 16 ode_b3, 17 dec_W1, 18 dec_b1, 19 dec_W2, 20 dec_b2
  const float* values = (const float*)d_in[1];
  const float* maskp  = (const float*)d_in[2];
  const int*   seqln  = (const int*)d_in[3];
  const float* eps    = (const float*)d_in[4];
  const float* W_ih   = (const float*)d_in[5];
  const float* W_hh   = (const float*)d_in[6];
  const float* b_ih   = (const float*)d_in[7];
  const float* b_hh   = (const float*)d_in[8];
  const float* W_z0   = (const float*)d_in[9];
  const float* b_z0   = (const float*)d_in[10];
  const float* oW1    = (const float*)d_in[11];
  const float* ob1    = (const float*)d_in[12];
  const float* oW2    = (const float*)d_in[13];
  const float* ob2    = (const float*)d_in[14];
  const float* oW3    = (const float*)d_in[15];
  const float* ob3    = (const float*)d_in[16];
  const float* dW1    = (const float*)d_in[17];
  const float* db1    = (const float*)d_in[18];
  const float* dW2    = (const float*)d_in[19];
  const float* db2    = (const float*)d_in[20];
  float* out = (float*)d_out;

  latent_ode_init_kernel<<<1, 32, 0, stream>>>(out);
  latent_ode_kernel<<<BB / 16, 32, 0, stream>>>(
      values, maskp, seqln, eps, W_ih, W_hh, b_ih, b_hh, W_z0, b_z0, oW1, ob1,
      oW2, ob2, oW3, ob3, dW1, db1, dW2, db2, out);
}
